// SpatialGate_27590869909618
// MI455X (gfx1250) — matrix operand order repaired and checked by reading, not compiled, not run on hardware
//
#include <hip/hip_runtime.h>
#include <math.h>

// Problem dims (from reference setup_inputs): B=16, C=256, H=W=128
#define B_   16
#define C_   256
#define H_   128
#define W_   128
#define HW_  (H_ * W_)          // 16384 = 2^14
#define CHW_ ((size_t)C_ * HW_) // 4194304 = 2^22

typedef __attribute__((ext_vector_type(16))) _Float16 v16h;
typedef __attribute__((ext_vector_type(8)))  float    v8f;

// ---------------------------------------------------------------------------
// Pass 1: channel reductions.  compress[b][0][hw] = max_c x, [b][1][hw] = mean_c y
// 65536 threads, each owns 4 consecutive (h,w) positions -> float4 coalesced.
// ---------------------------------------------------------------------------
__global__ void sg_reduce_kernel(const float* __restrict__ x,
                                 const float* __restrict__ y,
                                 float* __restrict__ compress) {
    int tid = blockIdx.x * blockDim.x + threadIdx.x;   // 0 .. 65535
    int b   = tid >> 12;                               // tid / (HW/4)
    int hw  = (tid & 4095) << 2;                       // 4-float chunk in image
    size_t base = (size_t)b * CHW_ + hw;

    float4 mx = *(const float4*)(x + base);
    float4 sy = *(const float4*)(y + base);
    for (int c = 1; c < C_; ++c) {
        float4 xv = *(const float4*)(x + base + (size_t)c * HW_);
        float4 yv = *(const float4*)(y + base + (size_t)c * HW_);
        mx.x = fmaxf(mx.x, xv.x);  mx.y = fmaxf(mx.y, xv.y);
        mx.z = fmaxf(mx.z, xv.z);  mx.w = fmaxf(mx.w, xv.w);
        sy.x += yv.x;  sy.y += yv.y;  sy.z += yv.z;  sy.w += yv.w;
    }
    const float inv = 1.0f / (float)C_;
    float4 mn = {sy.x * inv, sy.y * inv, sy.z * inv, sy.w * inv};
    float* cm = compress + (size_t)b * 2 * HW_ + hw;   // [B][2][HW]
    *(float4*)(cm)        = mx;   // channel 0: max(x)
    *(float4*)(cm + HW_)  = mn;   // channel 1: mean(y)
}

// ---------------------------------------------------------------------------
// Pass 2: 7x7 SAME conv (2->1 ch) + bias + sigmoid via v_wmma_f32_16x16x32_f16.
// One wave per strip of 16 consecutive w-positions in one row h.
// im2col K = 2*7*7 = 98, padded to 128 -> 4 WMMAs with f32 accumulation.
//   A (16x32 f16): rows = 16 pixels of the strip, per-lane gathered per the
//                  documented 16-bit A layout (lane<16: K half 0/16.., lane>=16:
//                  K half 8/24..).
//   B (32x16 f16): weights, identical in every N column (same k order as A: lane half 0: K 0-7,16-23,
//                  lane half 1: K 8-15,24-31).
// Then every D column holds the same 16 outputs; extract via the documented
// C/D layout: lanes 0-15 have M=0..7 in c[0..7], lanes 16-31 have M=8..15.
// ---------------------------------------------------------------------------
__global__ void sg_conv_wmma_kernel(const float* __restrict__ compress,
                                    const float* __restrict__ wgt,   // 98 floats (1,2,7,7)
                                    const float* __restrict__ bias,  // 1 float
                                    float* __restrict__ scale) {
    int wave = (blockIdx.x * blockDim.x + threadIdx.x) >> 5;  // strip id, 0..16383
    int lane = threadIdx.x & 31;

    int p0  = wave << 4;            // first pixel (flat over B*HW)
    int b   = p0 >> 14;             // / HW
    int rem = p0 & (HW_ - 1);
    int h   = rem >> 7;             // / W
    int w0  = rem & (W_ - 1);

    int m     = lane & 15;                // A-matrix row = pixel index in strip
    int khalf = (lane < 16) ? 0 : 8;      // A-layout K half for this lane
    int kbase = (lane < 16) ? 0 : 16;     // unused: B uses khalf like A
    const float* img = compress + (size_t)b * 2 * HW_;

    v8f acc = {};
#pragma unroll
    for (int t = 0; t < 4; ++t) {
        v16h afrag{};
        v16h bfrag{};
#pragma unroll
        for (int e = 0; e < 16; ++e) {
            int r = e >> 1;
            // --- A element k (im2col) per 16-bit A 16x32 layout ---
            int ka = t * 32 + ((r < 4) ? (khalf + 2 * r + (e & 1))
                                       : (16 + khalf + 2 * (r - 4) + (e & 1)));
            // --- B element k (weights) per 16-bit B 32x16 layout ---
            int kb = t * 32 + ((r < 4) ? (khalf + 2 * r + (e & 1))
                                       : (16 + khalf + 2 * (r - 4) + (e & 1)));

            float av = 0.0f;
            if (ka < 98) {
                int ci = ka / 49;
                int r2 = ka - ci * 49;
                int kh = r2 / 7;
                int kw = r2 - kh * 7;
                int hh = h + kh - 3;
                int ww = w0 + m + kw - 3;
                if (hh >= 0 && hh < H_ && ww >= 0 && ww < W_)
                    av = img[(size_t)ci * HW_ + hh * W_ + ww];
            }
            float bv = (kb < 98) ? wgt[kb] : 0.0f;
            afrag[e] = (_Float16)av;
            bfrag[e] = (_Float16)bv;
        }
        // D = A x B + C   (f32 accumulate)
        acc = __builtin_amdgcn_wmma_f32_16x16x32_f16(
            /*neg_a=*/false, afrag, /*neg_b=*/false, bfrag,
            /*c_mod=*/(short)0, acc, /*reuse_a=*/false, /*reuse_b=*/false);
    }

    float bb = bias[0];
    if (lane < 8) {
        float v = acc[lane] + bb;                       // pixel p0+lane (M=lane)
        scale[p0 + lane] = 1.0f / (1.0f + __expf(-v));
    } else if (lane >= 16 && lane < 24) {
        float v = acc[lane - 16] + bb;                  // pixel p0+8+(lane-16)
        scale[p0 + 8 + (lane - 16)] = 1.0f / (1.0f + __expf(-v));
    }
}

// ---------------------------------------------------------------------------
// Pass 3: out[b,c,h,w] = x[b,c,h,w] * scale[b,h,w]   (float4 streaming)
// ---------------------------------------------------------------------------
__global__ void sg_apply_kernel(const float* __restrict__ x,
                                const float* __restrict__ scale,
                                float* __restrict__ out) {
    size_t i  = (size_t)blockIdx.x * blockDim.x + threadIdx.x;
    size_t i4 = i << 2;
    int b  = (int)(i4 >> 22);          // / (C*HW)
    int hw = (int)(i4 & (HW_ - 1));    // % HW  (HW divides C*HW)
    float4 xv = *(const float4*)(x + i4);
    float4 sv = *(const float4*)(scale + (size_t)b * HW_ + hw);
    float4 o  = {xv.x * sv.x, xv.y * sv.y, xv.z * sv.z, xv.w * sv.w};
    *(float4*)(out + i4) = o;
}

// ---------------------------------------------------------------------------
extern "C" void kernel_launch(void* const* d_in, const int* in_sizes, int n_in,
                              void* d_out, int out_size, void* d_ws, size_t ws_size,
                              hipStream_t stream) {
    const float* x    = (const float*)d_in[0];
    const float* y    = (const float*)d_in[1];
    const float* w    = (const float*)d_in[2];   // 98 floats
    const float* bias = (const float*)d_in[3];   // 1 float
    float* out = (float*)d_out;

    // Workspace layout: compress [B][2][HW] (2 MB) | scale [B][HW] (1 MB)
    float* compress = (float*)d_ws;
    float* scale    = compress + (size_t)2 * B_ * HW_;

    // Pass 1: 65536 threads -> 256 blocks x 256
    sg_reduce_kernel<<<(B_ * HW_ / 4) / 256, 256, 0, stream>>>(x, y, compress);

    // Pass 2: 16384 strips, 8 waves per 256-thread block -> 2048 blocks
    sg_conv_wmma_kernel<<<(B_ * HW_ / 16) / 8, 256, 0, stream>>>(compress, w, bias, scale);

    // Pass 3: 16M float4 threads -> 65536 blocks x 256
    sg_apply_kernel<<<(unsigned)(((size_t)B_ * CHW_ / 4) / 256), 256, 0, stream>>>(x, scale, out);
}
